// InstanceLoss_80547816669829
// MI455X (gfx1250) — matrix operand order repaired and checked by reading, not compiled, not run on hardware
//
#include <hip/hip_runtime.h>
#include <math.h>

// CDNA5 / gfx1250: wave32, bf16x3 split-precision GEMM on V_WMMA_F32_16X16X32_BF16.
typedef __attribute__((ext_vector_type(16))) __bf16 v16bf;
typedef __attribute__((ext_vector_type(8)))  float  v8f;

#define N_ROWS 8192
#define DIM    256
#define EPSF   1e-6f
#define TILE   128           // block tile: 128x128 output
#define KC     32            // K chunk staged in LDS (= WMMA K depth)
#define LDB    40            // padded LDS row stride in bf16 elements (80 B)

// ---- bf16 round-to-nearest-even helpers (bit-exact, integer path) ----
static __device__ inline unsigned short f2bf_rne(float x) {
    unsigned u = __float_as_uint(x);
    unsigned r = u + 0x7FFFu + ((u >> 16) & 1u);
    return (unsigned short)(r >> 16);
}
static __device__ inline float bf2f(unsigned short h) {
    return __uint_as_float(((unsigned)h) << 16);
}

union FragU {
    v16bf v;
    uint4 q[2];
};

// ---------------- Kernel 1: per-row sum and squared-norm (fp32 exact) ----------------
__global__ __launch_bounds__(256) void rowstats_kernel(const float* __restrict__ E,
                                                       float* __restrict__ sqn,
                                                       float* __restrict__ rsum) {
    int row  = blockIdx.x * 8 + (threadIdx.x >> 5);
    int lane = threadIdx.x & 31;
    const float* p = E + (size_t)row * DIM;
    float s = 0.f, q = 0.f;
    #pragma unroll
    for (int k = lane; k < DIM; k += 32) {
        float x = p[k];
        s += x;
        q += x * x;
    }
    #pragma unroll
    for (int off = 16; off > 0; off >>= 1) {
        s += __shfl_down(s, off, 32);
        q += __shfl_down(q, off, 32);
    }
    if (lane == 0) { rsum[row] = s; sqn[row] = q; }
}

// ---------------- Kernel 2: bf16x3 WMMA gram + fused distance/mask/reduce ----------------
__global__ __launch_bounds__(256) void gram_dist_kernel(const float* __restrict__ E,
                                                        const int*   __restrict__ lab,
                                                        const float* __restrict__ sqn,
                                                        const float* __restrict__ rsum,
                                                        float* __restrict__ partials) {
    __shared__ __align__(16) unsigned short sAhi[TILE * LDB];   // 10240 B each
    __shared__ __align__(16) unsigned short sAlo[TILE * LDB];
    __shared__ __align__(16) unsigned short sBhi[TILE * LDB];
    __shared__ __align__(16) unsigned short sBlo[TILE * LDB];
    __shared__ float sred[256];

    const int iBase = blockIdx.y * TILE;
    const int jBase = blockIdx.x * TILE;

    const int tid  = threadIdx.x;
    const int wave = tid >> 5;        // 0..7: owns rows [16*wave, 16*wave+16)
    const int lane = tid & 31;
    const int half = lane >> 4;       // K-half selector in 16-bit WMMA operand layouts
    const int mr   = lane & 15;

    v8f acc[8] = {};                  // 8 subtiles of 16x16 across 128 columns

    const unsigned short* pAh = &sAhi[(wave * 16 + mr) * LDB];
    const unsigned short* pAl = &sAlo[(wave * 16 + mr) * LDB];

    for (int kc = 0; kc < DIM; kc += KC) {
        // Stage fp32 -> bf16 hi/lo into LDS (coalesced float4 global reads).
        #pragma unroll
        for (int it = 0; it < 4; ++it) {
            int idx = tid + it * 256;            // 0..1023 over 128 rows x 8 float4
            int row = idx >> 3;
            int c4  = (idx & 7) << 2;
            float4 va = *(const float4*)&E[(size_t)(iBase + row) * DIM + kc + c4];
            float4 vb = *(const float4*)&E[(size_t)(jBase + row) * DIM + kc + c4];
            float ax[4] = {va.x, va.y, va.z, va.w};
            float bx[4] = {vb.x, vb.y, vb.z, vb.w};
            unsigned ah[2] = {0, 0}, al[2] = {0, 0}, bh[2] = {0, 0}, bl[2] = {0, 0};
            #pragma unroll
            for (int e = 0; e < 4; ++e) {
                unsigned short h, l;
                h = f2bf_rne(ax[e]);
                l = f2bf_rne(ax[e] - bf2f(h));
                ah[e >> 1] |= ((unsigned)h) << ((e & 1) * 16);
                al[e >> 1] |= ((unsigned)l) << ((e & 1) * 16);
                h = f2bf_rne(bx[e]);
                l = f2bf_rne(bx[e] - bf2f(h));
                bh[e >> 1] |= ((unsigned)h) << ((e & 1) * 16);
                bl[e >> 1] |= ((unsigned)l) << ((e & 1) * 16);
            }
            *(uint2*)&sAhi[row * LDB + c4] = make_uint2(ah[0], ah[1]);
            *(uint2*)&sAlo[row * LDB + c4] = make_uint2(al[0], al[1]);
            *(uint2*)&sBhi[row * LDB + c4] = make_uint2(bh[0], bh[1]);
            *(uint2*)&sBlo[row * LDB + c4] = make_uint2(bl[0], bl[1]);
        }
        // Prefetch next K chunk into the cache hierarchy (global_prefetch_b8).
        if (kc + KC < DIM) {
            __builtin_prefetch(&E[(size_t)(iBase + (tid >> 1)) * DIM + kc + KC + (tid & 1) * 16], 0, 0);
        }
        __syncthreads();

        // A fragment (16-bit A 16x32): lane(m,h) -> 16B at K-elem 8h and 16+8h.
        FragU a_hi, a_lo;
        a_hi.q[0] = *(const uint4*)(pAh + 8 * half);
        a_hi.q[1] = *(const uint4*)(pAh + 16 + 8 * half);
        a_lo.q[0] = *(const uint4*)(pAl + 8 * half);
        a_lo.q[1] = *(const uint4*)(pAl + 16 + 8 * half);

        #pragma unroll
        for (int c = 0; c < 8; ++c) {
            // B fragment (16-bit B 32x16): lane(n,h) -> 16B at K-elem 8h and 16+8h.
            const unsigned short* pBh = &sBhi[(c * 16 + mr) * LDB + 8 * half];
            const unsigned short* pBl = &sBlo[(c * 16 + mr) * LDB + 8 * half];
            FragU b_hi, b_lo;
            b_hi.q[0] = *(const uint4*)(pBh);
            b_hi.q[1] = *(const uint4*)(pBh + 16);
            b_lo.q[0] = *(const uint4*)(pBl);
            b_lo.q[1] = *(const uint4*)(pBl + 16);
            // bf16x3: hi*hi + lo*hi + hi*lo  (lo*lo dropped; ~2^-16 relative)
            acc[c] = __builtin_amdgcn_wmma_f32_16x16x32_bf16(
                false, a_hi.v, false, b_hi.v, (short)0, acc[c], false, false);
            acc[c] = __builtin_amdgcn_wmma_f32_16x16x32_bf16(
                false, a_lo.v, false, b_hi.v, (short)0, acc[c], false, false);
            acc[c] = __builtin_amdgcn_wmma_f32_16x16x32_bf16(
                false, a_hi.v, false, b_lo.v, (short)0, acc[c], false, false);
        }
        __syncthreads();
    }

    // Epilogue: C/D 16x16 f32 layout -> lane holds (M = v + 8*half, N = mr), v = 0..7.
    const int i0 = iBase + wave * 16 + 8 * half;
    float si_sq[8], si_rs[8];
    int   si_lb[8];
    #pragma unroll
    for (int v = 0; v < 8; ++v) {
        si_sq[v] = sqn[i0 + v];
        si_rs[v] = rsum[i0 + v];
        si_lb[v] = lab[i0 + v];
    }

    float lsum = 0.f;
    const float cterm = (float)DIM * EPSF * EPSF;
    #pragma unroll
    for (int c = 0; c < 8; ++c) {
        int j = jBase + c * 16 + mr;
        float sjq = sqn[j];
        float sjr = rsum[j];
        int   sjl = lab[j];
        #pragma unroll
        for (int v = 0; v < 8; ++v) {
            float g  = acc[c][v];
            float sq = si_sq[v] + sjq - 2.0f * g
                     + 2.0f * EPSF * (si_rs[v] - sjr) + cterm;
            float d  = sqrtf(fmaxf(sq, 0.0f));
            lsum += (si_lb[v] == sjl) ? d : 0.0f;
        }
    }

    // Deterministic block tree reduction.
    sred[tid] = lsum;
    __syncthreads();
    #pragma unroll
    for (int s = 128; s > 0; s >>= 1) {
        if (tid < s) sred[tid] += sred[tid + s];
        __syncthreads();
    }
    if (tid == 0) partials[blockIdx.y * gridDim.x + blockIdx.x] = sred[0];
}

// ---------------- Kernel 3: final reduction + scale ----------------
__global__ __launch_bounds__(256) void finalize_kernel(const float* __restrict__ partials,
                                                       int n, float* __restrict__ out) {
    __shared__ float sred[256];
    float s = 0.f;
    for (int i = threadIdx.x; i < n; i += 256) s += partials[i];
    sred[threadIdx.x] = s;
    __syncthreads();
    #pragma unroll
    for (int k = 128; k > 0; k >>= 1) {
        if (threadIdx.x < k) sred[threadIdx.x] += sred[threadIdx.x + k];
        __syncthreads();
    }
    if (threadIdx.x == 0)
        out[0] = sred[0] * (1.0f / ((float)N_ROWS * (float)(N_ROWS - 1)));
}

extern "C" void kernel_launch(void* const* d_in, const int* in_sizes, int n_in,
                              void* d_out, int out_size, void* d_ws, size_t ws_size,
                              hipStream_t stream) {
    (void)in_sizes; (void)n_in; (void)out_size; (void)ws_size;
    const float* E   = (const float*)d_in[0];   // (8192, 256) fp32
    const int*   lab = (const int*)d_in[1];     // (8192,) int32
    float*       out = (float*)d_out;           // scalar fp32

    float* sqn      = (float*)d_ws;             // 8192 floats
    float* rsum     = sqn + N_ROWS;             // 8192 floats
    float* partials = rsum + N_ROWS;            // 4096 floats

    rowstats_kernel<<<N_ROWS / 8, 256, 0, stream>>>(E, sqn, rsum);

    dim3 grid(N_ROWS / TILE, N_ROWS / TILE);    // 64 x 64 = 4096 blocks
    gram_dist_kernel<<<grid, 256, 0, stream>>>(E, lab, sqn, rsum, partials);

    finalize_kernel<<<1, 256, 0, stream>>>(partials, N_ROWS / TILE * (N_ROWS / TILE), out);
}
